// GATNet_74406013436132
// MI455X (gfx1250) — matrix operand order repaired and checked by reading, not compiled, not run on hardware
//
#include <hip/hip_runtime.h>
#include <hip/hip_bf16.h>
#include <math.h>

// ---------------------------------------------------------------------------
// CDNA5 (gfx1250) GAT 3-layer forward.
//  - Activations converted once/layer to bf16; weights pre-packed into the
//    WMMA B-fragment dword layout.
//  - GEMM templated on (K, Nc): fully unrolled, branch-free inner loop of
//    immediate-offset loads + v_wmma_f32_16x16x32_bf16.
//  - Segment softmax + scatter aggregation via L2-resident atomics.
// ---------------------------------------------------------------------------

typedef __attribute__((ext_vector_type(16))) __bf16 v16bf;
typedef __attribute__((ext_vector_type(8)))  float  v8f;

// float -> bf16 (round to nearest even), low 16 bits (prep passes only)
__device__ __forceinline__ unsigned f2bf(float f) {
    union { float f; unsigned u; } x; x.f = f;
    unsigned u = x.u;
    return ((u + 0x7FFFu + ((u >> 16) & 1u)) >> 16) & 0xFFFFu;
}

// order-preserving float <-> uint encoding for atomicMax over signed floats
__device__ __forceinline__ unsigned f32_ord(float f) {
    union { float f; int i; unsigned u; } x; x.f = f;
    return (x.i >= 0) ? (x.u | 0x80000000u) : ~x.u;
}
__device__ __forceinline__ float ord_f32(unsigned k) {
    union { float f; unsigned u; } x;
    x.u = (k & 0x80000000u) ? (k & 0x7FFFFFFFu) : ~k;
    return x.f;
}

union Frag { v16bf v; unsigned u[8]; };

// ---------------------------------------------------------------------------
// prep passes
// ---------------------------------------------------------------------------
__global__ void f32_to_bf16(const float* __restrict__ X,
                            unsigned short* __restrict__ Y, size_t n) {
    const size_t i = (size_t)blockIdx.x * blockDim.x + threadIdx.x;
    if (i < n) Y[i] = (unsigned short)f2bf(X[i]);
}

// Bp[(k/2)*Nc + col] = { bf16(W[k][col]), bf16(W[k+1][col]) }  (lo, hi)
__global__ void prep_weight(const float* __restrict__ W, unsigned* __restrict__ Bp,
                            int K, int Nc) {
    const int i = blockIdx.x * blockDim.x + threadIdx.x;
    if (i >= (K / 2) * Nc) return;
    const int kp = i / Nc, col = i % Nc;
    Bp[i] = f2bf(W[(size_t)(2 * kp) * Nc + col]) |
            (f2bf(W[(size_t)(2 * kp + 1) * Nc + col]) << 16);
}

// ---------------------------------------------------------------------------
// C[M,Nc] = A[M,K] @ B[K,Nc]; A bf16 row-major, Bp packed (see prep_weight),
// C fp32 row-major. One wave -> 16 x (16*NSUB) tile. Branch-free, fully
// unrolled inner loop: OOB rows/cols are clamped (their results are simply
// never stored), so EXEC stays all-ones and all offsets are immediates.
// grid: (ceil(ceil(M/16)/8), Nc/64 or 1); block 256 (8 waves).
// ---------------------------------------------------------------------------
template <int K, int Nc>
__global__ void wmma_gemm_t(const unsigned short* __restrict__ A,
                            const unsigned* __restrict__ Bp,
                            float* __restrict__ C, int M) {
    constexpr int NSUB = (Nc >= 64) ? 4 : (Nc + 15) / 16;

    const int lane  = threadIdx.x & 31;
    const int wave  = threadIdx.x >> 5;
    const int mTile = blockIdx.x * (blockDim.x >> 5) + wave;
    const int n0    = blockIdx.y * 64;
    if (mTile * 16 >= M) return;                 // wave-uniform exit

    const int hi      = lane >> 4;               // 0: lanes 0-15, 1: lanes 16-31
    const int colBase = lane & 15;
    const int rowA    = min(mTile * 16 + colBase, M - 1);   // clamped A row
    const int koffA   = hi * 16;                 // A K sub-offset (elements)
    const int kpB     = hi * 8;                  // B K sub-offset (row pairs)

    // loop-invariant bases; all per-iteration accesses use immediate offsets
    const uint4* aBase = reinterpret_cast<const uint4*>(
        A + (size_t)rowA * K + koffA);
    const unsigned* bBase[NSUB];
    #pragma unroll
    for (int t = 0; t < NSUB; ++t) {
        const int col = min(n0 + t * 16 + colBase, Nc - 1); // clamped B col
        bBase[t] = Bp + (size_t)kpB * Nc + col;
    }

    // prefetch A rows two m-tiles ahead (same lane pattern, +32 rows)
    __builtin_prefetch(aBase + (size_t)4 * K, 0, 2);

    v8f acc[NSUB] = {};

    #pragma unroll
    for (int kb = 0; kb < K; kb += 32) {
        Frag a;
        const uint4 lo = aBase[(kb >> 3) + 0];   // K = kb + koffA + 0..7
        const uint4 hh = aBase[(kb >> 3) + 1];   // K = kb + koffA + 8..15
        a.u[0] = lo.x; a.u[1] = lo.y; a.u[2] = lo.z; a.u[3] = lo.w;
        a.u[4] = hh.x; a.u[5] = hh.y; a.u[6] = hh.z; a.u[7] = hh.w;

        #pragma unroll
        for (int t = 0; t < NSUB; ++t) {
            Frag b;
            #pragma unroll
            for (int j = 0; j < 8; ++j)          // rows 2j, 2j+1 of this col
                b.u[j] = bBase[t][(size_t)((kb >> 1) + j) * Nc];
            acc[t] = __builtin_amdgcn_wmma_f32_16x16x32_bf16(
                false, a.v, false, b.v, (short)0, acc[t], false, false);
        }
    }

    // store C (C/D layout: VGPR j -> row j + 8*hi, col = lane&15)
    #pragma unroll
    for (int t = 0; t < NSUB; ++t) {
        const int col = n0 + t * 16 + colBase;
        if (col >= Nc) continue;                 // only possible when Nc=40
        #pragma unroll
        for (int j = 0; j < 8; ++j) {
            const int r = mTile * 16 + j + hi * 8;
            if (r < M) C[(size_t)r * Nc + col] = acc[t][j];
        }
    }
}

// ---------------------------------------------------------------------------
// al_s[n,h] = sum_c h[n, h*C+c] * a_s[h*C+c]  (one wave per node)
// ---------------------------------------------------------------------------
__global__ void compute_al(const float* __restrict__ h,
                           const float* __restrict__ a_s,
                           const float* __restrict__ a_d,
                           float* __restrict__ alS, float* __restrict__ alD,
                           int n, int W, int H) {
    const int lane = threadIdx.x & 31;
    const int node = (int)((blockIdx.x * blockDim.x + threadIdx.x) >> 5);
    if (node >= n) return;
    const int C = W / H;
    const float* hp = h + (size_t)node * W;
    for (int head = 0; head < H; ++head) {
        float ss = 0.f, sd = 0.f;
        for (int c = lane; c < C; c += 32) {
            const float hv = hp[head * C + c];
            ss += hv * a_s[head * C + c];
            sd += hv * a_d[head * C + c];
        }
        #pragma unroll
        for (int off = 16; off > 0; off >>= 1) {
            ss += __shfl_xor(ss, off, 32);
            sd += __shfl_xor(sd, off, 32);
        }
        if (lane == 0) {
            alS[(size_t)node * H + head] = ss;
            alD[(size_t)node * H + head] = sd;
        }
    }
}

__device__ __forceinline__ void edge_sd(const int* src, const int* dst,
                                        int e, int E, int& s, int& d) {
    if (e < E) { s = src[e]; d = dst[e]; }
    else       { s = e - E; d = e - E; }          // self loops
}

// pass 1: e = leaky_relu(alS[s]+alD[d]); segment max via ordered-uint atomicMax
__global__ void edge_logits_max(const int* __restrict__ src, const int* __restrict__ dst,
                                const float* __restrict__ alS, const float* __restrict__ alD,
                                unsigned* __restrict__ emax, int E, int n, int H) {
    const int e = blockIdx.x * blockDim.x + threadIdx.x;
    if (e >= E + n) return;
    int s, d; edge_sd(src, dst, e, E, s, d);
    for (int h = 0; h < H; ++h) {
        float v = alS[(size_t)s * H + h] + alD[(size_t)d * H + h];
        v = v > 0.f ? v : 0.2f * v;
        atomicMax(&emax[(size_t)d * H + h], f32_ord(v));
    }
}

// pass 2: ex = exp(e - emax[d]); den[d] += ex
__global__ void edge_softmax_num(const int* __restrict__ src, const int* __restrict__ dst,
                                 const float* __restrict__ alS, const float* __restrict__ alD,
                                 const unsigned* __restrict__ emax,
                                 float* __restrict__ ex, float* __restrict__ den,
                                 int E, int n, int H) {
    const int e = blockIdx.x * blockDim.x + threadIdx.x;
    if (e >= E + n) return;
    int s, d; edge_sd(src, dst, e, E, s, d);
    for (int h = 0; h < H; ++h) {
        float v = alS[(size_t)s * H + h] + alD[(size_t)d * H + h];
        v = v > 0.f ? v : 0.2f * v;
        const float m  = ord_f32(emax[(size_t)d * H + h]);
        const float xv = __expf(v - m);
        ex[(size_t)e * H + h] = xv;
        atomicAdd(&den[(size_t)d * H + h], xv);
    }
}

// pass 3: out[d,:] += h[s,:] * alpha   (one wave per edge, coalesced channels)
__global__ void edge_scatter(const int* __restrict__ src, const int* __restrict__ dst,
                             const float* __restrict__ h,
                             const float* __restrict__ ex, const float* __restrict__ den,
                             float* __restrict__ out, int E, int n, int H, int W) {
    const int lane = threadIdx.x & 31;
    const int e    = (int)((blockIdx.x * blockDim.x + threadIdx.x) >> 5);
    if (e >= E + n) return;
    int s, d; edge_sd(src, dst, e, E, s, d);
    const int C = W / H;
    const float a0 = ex[(size_t)e * H] / den[(size_t)d * H];
    const float a1 = (H > 1) ? ex[(size_t)e * H + 1] / den[(size_t)d * H + 1] : a0;
    for (int c = lane; c < W; c += 32) {
        const float a = (c < C) ? a0 : a1;
        atomicAdd(&out[(size_t)d * W + c], h[(size_t)s * W + c] * a);
    }
}

// y = relu(((x + bias) - m) * rsqrt(v+eps) * g + be), in place
__global__ void bias_bn_relu(float* __restrict__ x, const float* __restrict__ b,
                             const float* __restrict__ g, const float* __restrict__ be,
                             const float* __restrict__ m, const float* __restrict__ v,
                             int n, int W) {
    const size_t i = (size_t)blockIdx.x * blockDim.x + threadIdx.x;
    if (i >= (size_t)n * W) return;
    const int c = (int)(i % W);
    float y = x[i] + b[c];
    y = (y - m[c]) * rsqrtf(v[c] + 1e-5f) * g[c] + be[c];
    x[i] = y > 0.f ? y : 0.f;
}

// out = log_softmax(out + b), one wave per node, W = 40
__global__ void bias_logsoftmax(float* __restrict__ out, const float* __restrict__ b,
                                int n, int W) {
    const int lane = threadIdx.x & 31;
    const int node = (int)((blockIdx.x * blockDim.x + threadIdx.x) >> 5);
    if (node >= n) return;
    float* p = out + (size_t)node * W;
    float x0 = 0.f, x1 = 0.f, v0 = -INFINITY, v1 = -INFINITY;
    const int c1 = lane + 32;
    if (lane < W) { x0 = p[lane] + b[lane]; v0 = x0; }
    if (c1   < W) { x1 = p[c1]   + b[c1];   v1 = x1; }
    float mx = fmaxf(v0, v1);
    #pragma unroll
    for (int off = 16; off > 0; off >>= 1) mx = fmaxf(mx, __shfl_xor(mx, off, 32));
    float sum = 0.f;
    if (lane < W) sum += __expf(x0 - mx);
    if (c1   < W) sum += __expf(x1 - mx);
    #pragma unroll
    for (int off = 16; off > 0; off >>= 1) sum += __shfl_xor(sum, off, 32);
    const float lse = mx + __logf(sum);
    if (lane < W) p[lane] = x0 - lse;
    if (c1   < W) p[c1]   = x1 - lse;
}

__global__ void fill_f32(float* p, float v, size_t n) {
    const size_t i = (size_t)blockIdx.x * blockDim.x + threadIdx.x;
    if (i < n) p[i] = v;
}
__global__ void fill_u32(unsigned* p, unsigned v, size_t n) {
    const size_t i = (size_t)blockIdx.x * blockDim.x + threadIdx.x;
    if (i < n) p[i] = v;
}

// ---------------------------------------------------------------------------
extern "C" void kernel_launch(void* const* d_in, const int* in_sizes, int n_in,
                              void* d_out, int out_size, void* d_ws, size_t ws_size,
                              hipStream_t stream) {
    const float* x   = (const float*)d_in[0];
    const int*   src = (const int*)  d_in[1];
    const int*   dst = (const int*)  d_in[2];
    const float* W1  = (const float*)d_in[3];
    const float* as1 = (const float*)d_in[4];
    const float* ad1 = (const float*)d_in[5];
    const float* b1  = (const float*)d_in[6];
    const float* g1  = (const float*)d_in[7];
    const float* be1 = (const float*)d_in[8];
    const float* m1  = (const float*)d_in[9];
    const float* v1  = (const float*)d_in[10];
    const float* W2  = (const float*)d_in[11];
    const float* as2 = (const float*)d_in[12];
    const float* ad2 = (const float*)d_in[13];
    const float* b2  = (const float*)d_in[14];
    const float* g2  = (const float*)d_in[15];
    const float* be2 = (const float*)d_in[16];
    const float* m2  = (const float*)d_in[17];
    const float* v2  = (const float*)d_in[18];
    const float* W3  = (const float*)d_in[19];
    const float* as3 = (const float*)d_in[20];
    const float* ad3 = (const float*)d_in[21];
    const float* b3  = (const float*)d_in[22];
    float* out = (float*)d_out;

    const int n    = in_sizes[0] / 128;   // nodes
    const int E    = in_sizes[1];         // directed edges (pre self-loop)
    const int Etot = E + n;
    const int D    = 256, NCLS = 40;

    // workspace partition (256B aligned)
    char* p = (char*)d_ws;
    auto alloc = [&](size_t bytes) -> void* {
        void* r = (void*)p;
        p += (bytes + 255) & ~(size_t)255;
        return r;
    };
    float*          bufA  = (float*)         alloc((size_t)n * D * 4);     // h (projection)
    float*          bufB  = (float*)         alloc((size_t)n * D * 4);     // aggregated out
    float*          bufC  = (float*)         alloc((size_t)n * NCLS * 4);  // h layer 3
    unsigned short* Abf   = (unsigned short*)alloc((size_t)n * D * 2);     // bf16 activations
    unsigned*       Wp    = (unsigned*)      alloc((size_t)D * D / 2 * 4); // packed weights
    float*          alS   = (float*)         alloc((size_t)n * 2 * 4);
    float*          alD   = (float*)         alloc((size_t)n * 2 * 4);
    unsigned*       emax  = (unsigned*)      alloc((size_t)n * 2 * 4);
    float*          den   = (float*)         alloc((size_t)n * 2 * 4);
    float*          ex    = (float*)         alloc((size_t)Etot * 2 * 4);

    const int mtiles  = (n + 15) / 16;
    const int gemmBx  = (mtiles + 7) / 8;               // 8 waves per block
    const int nodeWav = (n + 7) / 8;                    // 1 wave/node blocks
    const int edgeThr = (Etot + 255) / 256;             // 1 thread/edge blocks
    const int edgeWav = (Etot + 7) / 8;                 // 1 wave/edge blocks
    const dim3 blk(256);

    auto gat_layer = [&](const float* hin, const float* Wt, const float* a_s,
                         const float* a_d, float* hproj, float* oagg,
                         int K, int W, int H) {
        f32_to_bf16<<<(int)(((size_t)n * K + 255) / 256), blk, 0, stream>>>(hin, Abf, (size_t)n * K);
        prep_weight<<<((K / 2) * W + 255) / 256, blk, 0, stream>>>(Wt, Wp, K, W);
        if (K == 128 && W == 256)
            wmma_gemm_t<128, 256><<<dim3(gemmBx, 4), blk, 0, stream>>>(Abf, Wp, hproj, n);
        else if (K == 256 && W == 256)
            wmma_gemm_t<256, 256><<<dim3(gemmBx, 4), blk, 0, stream>>>(Abf, Wp, hproj, n);
        else
            wmma_gemm_t<256, 40><<<dim3(gemmBx, 1), blk, 0, stream>>>(Abf, Wp, hproj, n);
        compute_al<<<nodeWav, blk, 0, stream>>>(hproj, a_s, a_d, alS, alD, n, W, H);
        fill_u32<<<(n * H + 255) / 256, blk, 0, stream>>>(emax, 0u, (size_t)n * H);
        fill_f32<<<(n * H + 255) / 256, blk, 0, stream>>>(den, 0.f, (size_t)n * H);
        fill_f32<<<(int)(((size_t)n * W + 255) / 256), blk, 0, stream>>>(oagg, 0.f, (size_t)n * W);
        edge_logits_max<<<edgeThr, blk, 0, stream>>>(src, dst, alS, alD, emax, E, n, H);
        edge_softmax_num<<<edgeThr, blk, 0, stream>>>(src, dst, alS, alD, emax, ex, den, E, n, H);
        edge_scatter<<<edgeWav, blk, 0, stream>>>(src, dst, hproj, ex, den, oagg, E, n, H, W);
    };

    // layer 1: 128 -> 256 (2 heads x 128), then bias+BN+ReLU
    gat_layer(x, W1, as1, ad1, bufA, bufB, 128, D, 2);
    bias_bn_relu<<<(int)(((size_t)n * D + 255) / 256), blk, 0, stream>>>(bufB, b1, g1, be1, m1, v1, n, D);

    // layer 2: 256 -> 256 (1 head), then bias+BN+ReLU
    gat_layer(bufB, W2, as2, ad2, bufA, bufB, D, D, 1);
    bias_bn_relu<<<(int)(((size_t)n * D + 255) / 256), blk, 0, stream>>>(bufB, b2, g2, be2, m2, v2, n, D);

    // layer 3: 256 -> 40 (1 head), aggregate into d_out, then +b3 & log_softmax
    gat_layer(bufB, W3, as3, ad3, bufC, out, D, NCLS, 1);
    bias_logsoftmax<<<nodeWav, blk, 0, stream>>>(out, b3, n, NCLS);

    (void)n_in; (void)out_size; (void)ws_size;
}
